// POCModelV2_60945585931023
// MI455X (gfx1250) — matrix operand order repaired and checked by reading, not compiled, not run on hardware
//
#include <hip/hip_runtime.h>

// ---------------------------------------------------------------------------
// CDNA5 (gfx1250) implementation: GCN encoders via bf16 WMMA GEMMs
// (v_wmma_f32_16x16x32_bf16, fp32 accumulate), atomic edge scatter for the
// graph propagation, fused attention+MLP head.
// Round 4: explicit register double-buffering in the GEMM K-loop (prefetch
// next A/B fragments before the current 4 WMMAs) + relaxed launch bounds so
// the allocator keeps both buffers live -> overlapped s_wait_loadcnt instead
// of a full drain before every WMMA.
// ---------------------------------------------------------------------------

typedef __attribute__((ext_vector_type(16))) __bf16 v16bf;
typedef __attribute__((ext_vector_type(8)))  float  v8f;

__device__ __forceinline__ unsigned short f2bf(float f) {
  union { float f; unsigned u; } x; x.f = f;
  unsigned r = x.u + 0x7FFFu + ((x.u >> 16) & 1u);   // round-to-nearest-even
  return (unsigned short)(r >> 16);
}

// ---------------- utility kernels ----------------
__global__ void fill_f32(float* __restrict__ p, float v, int n) {
  int i = blockIdx.x * blockDim.x + threadIdx.x;
  if (i < n) p[i] = v;
}

__global__ void deg_accum(const int* __restrict__ dst, int E, float* __restrict__ deg) {
  int e = blockIdx.x * blockDim.x + threadIdx.x;
  if (e < E) atomicAdd(&deg[dst[e]], 1.0f);
}

__global__ void rsqrt_inplace(float* __restrict__ p, int n) {
  int i = blockIdx.x * blockDim.x + threadIdx.x;
  if (i < n) { float d = p[i]; p[i] = (d > 0.0f) ? rsqrtf(d) : 0.0f; }
}

// ---------------- bf16 fragment packing ----------------
// A tile (16x32, MxK): per CDNA5 ISA 16-bit A layout.
// lane l: m = l&15, khalf = l>>4; element e: v=e>>1, lo=e&1,
// kbase = v<4 ? 2v : 16+2(v-4); k = kbase + lo + 8*khalf.
// Tile order: tile_id = mt*KT + kt; element addr = tile_id*512 + lane*16 + e.
__global__ void pack_a_bf16(const float* __restrict__ src, unsigned short* __restrict__ Ap,
                            int n, int K, int Kpad) {
  int idx = blockIdx.x * blockDim.x + threadIdx.x;
  int KT = Kpad >> 5;
  int total = (n >> 4) * KT * 512;
  if (idx >= total) return;
  int e    = idx & 15;
  int lane = (idx >> 4) & 31;
  int tile = idx >> 9;
  int kt = tile % KT;
  int mt = tile / KT;
  int v  = e >> 1;
  int kb = (v < 4) ? (v * 2) : (16 + (v - 4) * 2);
  int k  = kt * 32 + kb + (e & 1) + ((lane >> 4) << 3);
  int row = mt * 16 + (lane & 15);
  float val = (k < K) ? src[(size_t)row * K + k] : 0.0f;
  Ap[idx] = f2bf(val);
}

// B tile (32x16, KxN), N fixed = 128: lane l: ncol = l&15, khalf = l>>4;
// element e: k = (e&7) + 16*(e>>3) + 8*khalf. Tile order: tile_id = nt*KT + kt.
__global__ void pack_b_bf16(const float* __restrict__ src, unsigned short* __restrict__ Bp,
                            int K, int Kpad) {
  int idx = blockIdx.x * blockDim.x + threadIdx.x;
  int KT = Kpad >> 5;
  int total = 8 * KT * 512;               // NT = 8 (N = 128)
  if (idx >= total) return;
  int e    = idx & 15;
  int lane = (idx >> 4) & 31;
  int tile = idx >> 9;
  int kt = tile % KT;
  int nt = tile / KT;
  int k   = kt * 32 + (e & 7) + ((e >> 3) << 4) + ((lane >> 4) << 3);
  int col = nt * 16 + (lane & 15);
  float val = (k < K) ? src[(size_t)k * 128 + col] : 0.0f;
  Bp[idx] = f2bf(val);
}

// ---------------- WMMA GEMM, C[M x 128] = A[M x Kpad] @ B[Kpad x 128] ----------------
// One wave per (16-row tile, group of four 16-col tiles): A fragment reused x4.
// Explicit double-buffering: next K-step fragments are loaded before the
// current step's WMMAs are issued.
__global__ void __launch_bounds__(256, 1)
gemm_bf16_n128(const unsigned short* __restrict__ Ap,
               const unsigned short* __restrict__ Bp,
               float* __restrict__ C, const float* __restrict__ bias,
               int MT, int KT) {
  int wave = (int)((blockIdx.x * blockDim.x + threadIdx.x) >> 5);
  if (wave >= MT * 2) return;             // wave-uniform exit (EXEC stays all-ones)
  int lane = threadIdx.x & 31;
  int mt = wave >> 1;
  int ng = wave & 1;                      // n-group: tiles ng*4 .. ng*4+3

  v8f acc0 = {}; v8f acc1 = {}; v8f acc2 = {}; v8f acc3 = {};
  const size_t bTS = (size_t)KT * 512;    // stride between n-tile columns of B
  const unsigned short* aPtr = Ap + ((size_t)mt * KT) * 512 + lane * 16;
  const unsigned short* bPtr = Bp + ((size_t)(ng * 4) * KT) * 512 + lane * 16;

  v16bf a  = *(const v16bf*)aPtr;
  v16bf b0 = *(const v16bf*)(bPtr);
  v16bf b1 = *(const v16bf*)(bPtr + bTS);
  v16bf b2 = *(const v16bf*)(bPtr + 2 * bTS);
  v16bf b3 = *(const v16bf*)(bPtr + 3 * bTS);

  for (int kt = 0; kt < KT - 1; ++kt) {
    aPtr += 512;
    bPtr += 512;
    // prefetch next K-step into fresh registers
    v16bf an  = *(const v16bf*)aPtr;
    v16bf bn0 = *(const v16bf*)(bPtr);
    v16bf bn1 = *(const v16bf*)(bPtr + bTS);
    v16bf bn2 = *(const v16bf*)(bPtr + 2 * bTS);
    v16bf bn3 = *(const v16bf*)(bPtr + 3 * bTS);
    // consume current fragments
    acc0 = __builtin_amdgcn_wmma_f32_16x16x32_bf16(false, a, false, b0, (short)0, acc0, false, false);
    acc1 = __builtin_amdgcn_wmma_f32_16x16x32_bf16(false, a, false, b1, (short)0, acc1, false, false);
    acc2 = __builtin_amdgcn_wmma_f32_16x16x32_bf16(false, a, false, b2, (short)0, acc2, false, false);
    acc3 = __builtin_amdgcn_wmma_f32_16x16x32_bf16(false, a, false, b3, (short)0, acc3, false, false);
    a = an; b0 = bn0; b1 = bn1; b2 = bn2; b3 = bn3;
  }
  acc0 = __builtin_amdgcn_wmma_f32_16x16x32_bf16(false, a, false, b0, (short)0, acc0, false, false);
  acc1 = __builtin_amdgcn_wmma_f32_16x16x32_bf16(false, a, false, b1, (short)0, acc1, false, false);
  acc2 = __builtin_amdgcn_wmma_f32_16x16x32_bf16(false, a, false, b2, (short)0, acc2, false, false);
  acc3 = __builtin_amdgcn_wmma_f32_16x16x32_bf16(false, a, false, b3, (short)0, acc3, false, false);

  // C layout: VGPR r -> row mt*16 + r + 8*(lane>>4), col nt*16 + (lane&15)
  int rbase = mt * 16 + ((lane >> 4) << 3);
  int nl = lane & 15;
  int col0 = (ng * 4 + 0) * 16 + nl;
  int col1 = (ng * 4 + 1) * 16 + nl;
  int col2 = (ng * 4 + 2) * 16 + nl;
  int col3 = (ng * 4 + 3) * 16 + nl;
  float bb0 = bias ? bias[col0] : 0.0f;
  float bb1 = bias ? bias[col1] : 0.0f;
  float bb2 = bias ? bias[col2] : 0.0f;
  float bb3 = bias ? bias[col3] : 0.0f;
#pragma unroll
  for (int r = 0; r < 8; ++r) {
    size_t rowOff = (size_t)(rbase + r) * 128;
    C[rowOff + col0] = acc0[r] + bb0;
    C[rowOff + col1] = acc1[r] + bb1;
    C[rowOff + col2] = acc2[r] + bb2;
    C[rowOff + col3] = acc3[r] + bb3;
  }
}

// ---------------- GCN propagation ----------------
// acc[d] += h[s] * dinv[s]   (one wave per edge, 4 channels/lane)
__global__ void edge_scatter(const int* __restrict__ src, const int* __restrict__ dst,
                             int E, const float* __restrict__ h,
                             const float* __restrict__ dinv, float* __restrict__ acc) {
  int t = blockIdx.x * blockDim.x + threadIdx.x;
  int e = t >> 5;
  if (e >= E) return;
  int lane = t & 31;
  int s = src[e], d = dst[e];
  float w = dinv[s];
  float4 v = *(const float4*)(h + (size_t)s * 128 + lane * 4);
  float* a = acc + (size_t)d * 128 + lane * 4;
  atomicAdd(a + 0, v.x * w);
  atomicAdd(a + 1, v.y * w);
  atomicAdd(a + 2, v.z * w);
  atomicAdd(a + 3, v.w * w);
}

// out = relu( dinv[i]*(acc[i] + h[i]*dinv[i]) + b )   (self-loop folded in)
__global__ void gcn_finalize(const float* __restrict__ acc, const float* __restrict__ h,
                             const float* __restrict__ dinv, const float* __restrict__ bias,
                             float* __restrict__ out, int n) {
  int idx = blockIdx.x * blockDim.x + threadIdx.x;
  if (idx >= n * 128) return;
  int i = idx >> 7, c = idx & 127;
  float di = dinv[i];
  float v = di * (acc[idx] + h[idx] * di) + bias[c];
  out[idx] = fmaxf(v, 0.0f);
}

// ---------------- mean pooling ----------------
__global__ void pool_accum(const float* __restrict__ x, const int* __restrict__ batch,
                           float* __restrict__ pool, float* __restrict__ cnt, int n) {
  int idx = blockIdx.x * blockDim.x + threadIdx.x;
  if (idx >= n * 128) return;
  int i = idx >> 7, c = idx & 127;
  atomicAdd(&pool[(size_t)batch[i] * 128 + c], x[idx]);
  if (c == 0) atomicAdd(&cnt[batch[i]], 1.0f);
}

__global__ void pool_div(const float* __restrict__ pool, const float* __restrict__ cnt,
                         float* __restrict__ emb, int B) {
  int idx = blockIdx.x * blockDim.x + threadIdx.x;
  if (idx >= B * 128) return;
  emb[idx] = pool[idx] / fmaxf(cnt[idx >> 7], 1.0f);
}

// protEmb[C][B][H] -> confmat[(b*5+cf)][H]
__global__ void build_conf(const float* __restrict__ protEmb, float* __restrict__ confmat) {
  int idx = blockIdx.x * blockDim.x + threadIdx.x;
  if (idx >= 32 * 5 * 128) return;
  int h = idx & 127;
  int t = idx >> 7;
  int b = t / 5, cf = t % 5;
  confmat[idx] = protEmb[((size_t)cf * 32 + b) * 128 + h];
}

// ---------------- attention + MLP head (one block per batch element) ----------------
__global__ void __launch_bounds__(128)
attn_mlp(const float* __restrict__ q, const float* __restrict__ keys,
         const float* __restrict__ vals, const float* __restrict__ drugEmb,
         const float* __restrict__ W1, const float* __restrict__ b1,
         const float* __restrict__ W2, const float* __restrict__ b2,
         float* __restrict__ logits, float* __restrict__ attnOut) {
  int b = blockIdx.x, t = threadIdx.x;
  __shared__ float red[128];
  __shared__ float sc[5];
  __shared__ float attn[5];
  __shared__ float comb[256];

  float qv = q[(size_t)b * 128 + t];
  for (int cf = 0; cf < 5; ++cf) {
    red[t] = qv * keys[((size_t)b * 5 + cf) * 128 + t];
    __syncthreads();
    for (int s = 64; s > 0; s >>= 1) { if (t < s) red[t] += red[t + s]; __syncthreads(); }
    if (t == 0) sc[cf] = red[0] * 0.08838834764831845f;   // 1/sqrt(128)
    __syncthreads();
  }
  if (t == 0) {
    float mx = sc[0];
    for (int i = 1; i < 5; ++i) mx = fmaxf(mx, sc[i]);
    float ssum = 0.0f;
    for (int i = 0; i < 5; ++i) { attn[i] = expf(sc[i] - mx); ssum += attn[i]; }
    for (int i = 0; i < 5; ++i) attn[i] /= ssum;
  }
  __syncthreads();

  float pe = 0.0f;
  for (int cf = 0; cf < 5; ++cf) pe += attn[cf] * vals[((size_t)b * 5 + cf) * 128 + t];
  comb[t]       = drugEmb[(size_t)b * 128 + t];
  comb[128 + t] = pe;
  __syncthreads();

  float hsum = b1[t];
  for (int k = 0; k < 256; ++k) hsum += comb[k] * W1[(size_t)k * 128 + t];
  float hid = fmaxf(hsum, 0.0f);
  red[t] = hid * W2[t];
  __syncthreads();
  for (int s = 64; s > 0; s >>= 1) { if (t < s) red[t] += red[t + s]; __syncthreads(); }
  if (t == 0) logits[b] = red[0] + b2[0];
  if (t < 5) attnOut[b * 5 + t] = attn[t];
}

// ---------------------------------------------------------------------------
extern "C" void kernel_launch(void* const* d_in, const int* in_sizes, int n_in,
                              void* d_out, int out_size, void* d_ws, size_t ws_size,
                              hipStream_t stream) {
  const float* drug_x = (const float*)d_in[0];
  const float* prot_x = (const float*)d_in[1];
  const float* Wd0 = (const float*)d_in[2];  const float* bd0 = (const float*)d_in[3];
  const float* Wd1 = (const float*)d_in[4];  const float* bd1 = (const float*)d_in[5];
  const float* Wd2 = (const float*)d_in[6];  const float* bd2 = (const float*)d_in[7];
  const float* Wp0 = (const float*)d_in[8];  const float* bp0 = (const float*)d_in[9];
  const float* Wp1 = (const float*)d_in[10]; const float* bp1 = (const float*)d_in[11];
  const float* Wp2 = (const float*)d_in[12]; const float* bp2 = (const float*)d_in[13];
  const float* Wq  = (const float*)d_in[14]; const float* bq  = (const float*)d_in[15];
  const float* Wk  = (const float*)d_in[16]; const float* bk  = (const float*)d_in[17];
  const float* Wv  = (const float*)d_in[18]; const float* bv  = (const float*)d_in[19];
  const float* W1  = (const float*)d_in[20]; const float* b1  = (const float*)d_in[21];
  const float* W2  = (const float*)d_in[22]; const float* b2  = (const float*)d_in[23];
  const int* dEI    = (const int*)d_in[24];
  const int* pEI    = (const int*)d_in[25];
  const int* dBatch = (const int*)d_in[26];
  const int* pBatch = (const int*)d_in[27];
  (void)in_sizes; (void)n_in; (void)out_size; (void)ws_size;

  const int B = 32, C = 5, Nd = 1280, Ed = 5120, Np = 12800, Ep = 204800;
  const int Fd = 78, FdP = 96, Fp = 1280;

  // ---- workspace carve ----
  char* ws = (char*)d_ws; size_t off = 0;
  auto alloc = [&](size_t bytes) -> void* {
    void* p = ws + off; off = (off + bytes + 255) & ~(size_t)255; return p;
  };
  unsigned short* Ap   = (unsigned short*)alloc((size_t)Np * Fp * sizeof(unsigned short)); // 32.8 MB
  float* X    = (float*)alloc((size_t)Np * 128 * 4);
  float* Hb   = (float*)alloc((size_t)Np * 128 * 4);
  float* Acc  = (float*)alloc((size_t)Np * 128 * 4);
  float* Dinv = (float*)alloc((size_t)Np * 4);
  unsigned short* BpWd0 = (unsigned short*)alloc((size_t)3  * 8 * 512 * 2);
  unsigned short* BpWd1 = (unsigned short*)alloc((size_t)4  * 8 * 512 * 2);
  unsigned short* BpWd2 = (unsigned short*)alloc((size_t)4  * 8 * 512 * 2);
  unsigned short* BpWp0 = (unsigned short*)alloc((size_t)40 * 8 * 512 * 2);
  unsigned short* BpWp1 = (unsigned short*)alloc((size_t)4  * 8 * 512 * 2);
  unsigned short* BpWp2 = (unsigned short*)alloc((size_t)4  * 8 * 512 * 2);
  unsigned short* BpWq  = (unsigned short*)alloc((size_t)4  * 8 * 512 * 2);
  unsigned short* BpWk  = (unsigned short*)alloc((size_t)4  * 8 * 512 * 2);
  unsigned short* BpWv  = (unsigned short*)alloc((size_t)4  * 8 * 512 * 2);
  float* drugEmb = (float*)alloc((size_t)B * 128 * 4);
  float* protEmb = (float*)alloc((size_t)C * B * 128 * 4);
  float* confmat = (float*)alloc((size_t)B * C * 128 * 4);
  float* qbuf    = (float*)alloc((size_t)B * 128 * 4);
  float* keys    = (float*)alloc((size_t)B * C * 128 * 4);
  float* vals    = (float*)alloc((size_t)B * C * 128 * 4);
  float* pool    = (float*)alloc((size_t)B * 128 * 4);
  float* cnt     = (float*)alloc((size_t)B * 4);

  auto cdiv = [](int a, int b) { return (a + b - 1) / b; };

  // ---- pack all weight matrices to bf16 WMMA fragment layout (once) ----
  pack_b_bf16<<<cdiv(8 * 3  * 512, 256), 256, 0, stream>>>(Wd0, BpWd0, Fd,  FdP);
  pack_b_bf16<<<cdiv(8 * 4  * 512, 256), 256, 0, stream>>>(Wd1, BpWd1, 128, 128);
  pack_b_bf16<<<cdiv(8 * 4  * 512, 256), 256, 0, stream>>>(Wd2, BpWd2, 128, 128);
  pack_b_bf16<<<cdiv(8 * 40 * 512, 256), 256, 0, stream>>>(Wp0, BpWp0, Fp,  Fp);
  pack_b_bf16<<<cdiv(8 * 4  * 512, 256), 256, 0, stream>>>(Wp1, BpWp1, 128, 128);
  pack_b_bf16<<<cdiv(8 * 4  * 512, 256), 256, 0, stream>>>(Wp2, BpWp2, 128, 128);
  pack_b_bf16<<<cdiv(8 * 4  * 512, 256), 256, 0, stream>>>(Wq,  BpWq,  128, 128);
  pack_b_bf16<<<cdiv(8 * 4  * 512, 256), 256, 0, stream>>>(Wk,  BpWk,  128, 128);
  pack_b_bf16<<<cdiv(8 * 4  * 512, 256), 256, 0, stream>>>(Wv,  BpWv,  128, 128);

  auto compute_dinv = [&](const int* dst, int E, int n) {
    fill_f32<<<cdiv(n, 256), 256, 0, stream>>>(Dinv, 1.0f, n);   // self-loop degree
    deg_accum<<<cdiv(E, 256), 256, 0, stream>>>(dst, E, Dinv);
    rsqrt_inplace<<<cdiv(n, 256), 256, 0, stream>>>(Dinv, n);
  };

  auto gcn_layer = [&](const float* xin, int n, int K, int Kpad,
                       const unsigned short* Bw, const float* bias,
                       const int* src, const int* dst, int E, float* xout) {
    int KT = Kpad >> 5;
    int MT = n >> 4;
    pack_a_bf16<<<cdiv(MT * KT * 512, 256), 256, 0, stream>>>(xin, Ap, n, K, Kpad);
    gemm_bf16_n128<<<cdiv(MT * 2 * 32, 256), 256, 0, stream>>>(Ap, Bw, Hb, nullptr, MT, KT);
    fill_f32<<<cdiv(n * 128, 256), 256, 0, stream>>>(Acc, 0.0f, n * 128);
    edge_scatter<<<cdiv(E * 32, 256), 256, 0, stream>>>(src, dst, E, Hb, Dinv, Acc);
    gcn_finalize<<<cdiv(n * 128, 256), 256, 0, stream>>>(Acc, Hb, Dinv, bias, xout, n);
  };

  auto mean_pool = [&](const float* x, const int* batch, int n, float* emb) {
    fill_f32<<<cdiv(B * 128, 256), 256, 0, stream>>>(pool, 0.0f, B * 128);
    fill_f32<<<1, 256, 0, stream>>>(cnt, 0.0f, B);
    pool_accum<<<cdiv(n * 128, 256), 256, 0, stream>>>(x, batch, pool, cnt, n);
    pool_div<<<cdiv(B * 128, 256), 256, 0, stream>>>(pool, cnt, emb, B);
  };

  // ---- drug encoder ----
  compute_dinv(dEI + Ed, Ed, Nd);
  gcn_layer(drug_x, Nd, Fd, FdP, BpWd0, bd0, dEI, dEI + Ed, Ed, X);
  gcn_layer(X,      Nd, 128, 128, BpWd1, bd1, dEI, dEI + Ed, Ed, X);
  gcn_layer(X,      Nd, 128, 128, BpWd2, bd2, dEI, dEI + Ed, Ed, X);
  mean_pool(X, dBatch, Nd, drugEmb);

  // ---- protein encoder per conformation ----
  for (int cf = 0; cf < C; ++cf) {
    const int* pe = pEI + (size_t)cf * 2 * Ep;
    compute_dinv(pe + Ep, Ep, Np);
    gcn_layer(prot_x + (size_t)cf * Np * Fp, Np, Fp, Fp, BpWp0, bp0, pe, pe + Ep, Ep, X);
    gcn_layer(X, Np, 128, 128, BpWp1, bp1, pe, pe + Ep, Ep, X);
    gcn_layer(X, Np, 128, 128, BpWp2, bp2, pe, pe + Ep, Ep, X);
    mean_pool(X, pBatch, Np, protEmb + (size_t)cf * B * 128);
  }

  // ---- attention projections (WMMA) ----
  build_conf<<<cdiv(B * C * 128, 256), 256, 0, stream>>>(protEmb, confmat);
  {
    int MT = B / 16; // 2
    pack_a_bf16<<<cdiv(MT * 4 * 512, 256), 256, 0, stream>>>(drugEmb, Ap, B, 128, 128);
    gemm_bf16_n128<<<cdiv(MT * 2 * 32, 256), 256, 0, stream>>>(Ap, BpWq, qbuf, bq, MT, 4);
  }
  {
    int MT = (B * C) / 16; // 10
    pack_a_bf16<<<cdiv(MT * 4 * 512, 256), 256, 0, stream>>>(confmat, Ap, B * C, 128, 128);
    gemm_bf16_n128<<<cdiv(MT * 2 * 32, 256), 256, 0, stream>>>(Ap, BpWk, keys, bk, MT, 4);
    gemm_bf16_n128<<<cdiv(MT * 2 * 32, 256), 256, 0, stream>>>(Ap, BpWv, vals, bv, MT, 4);
  }

  // ---- attention + MLP head ----
  float* logits  = (float*)d_out;
  float* attnOut = (float*)d_out + 32;
  attn_mlp<<<B, 128, 0, stream>>>(qbuf, keys, vals, drugEmb, W1, b1, W2, b2, logits, attnOut);
}
